// My_HEAD_18253611008033
// MI455X (gfx1250) — compile-verified, matrix operand order checked by reading, not run on hardware
//
#include <hip/hip_runtime.h>

// ---------------------------------------------------------------------------
// out[n,k] = -||x_n - mu_k||^2 / (2*sum(std_k^2)+1e-8) - log(sum(std_k)) - 0.5*log(2pi)
//          = inv_denom[k] * (2*(x_n . mu_k) - ||x_n||^2) + beta[k]
// where beta[k] = -||mu_k||^2 * inv_denom[k] - log(sum std_k) - 0.5*log(2pi)
//
// GEMM core: split-bf16 (hi+lo) 3-product accumulation on v_wmma_f32_16x16x32_bf16.
// Wave tile 32x64 (8 accumulators): 24 WMMAs per 24 b128 loads per K-step.
// bf16 working set (36 MB) fits MI455X's 192 MB L2 -> no LDS staging.
// Workspace use: ~36.1 MB of d_ws.
// ---------------------------------------------------------------------------

typedef unsigned short u16;
typedef __attribute__((ext_vector_type(16))) __bf16 v16bf;
typedef __attribute__((ext_vector_type(8)))  __bf16 v8bf;
typedef __attribute__((ext_vector_type(8)))  float  v8f;

constexpr int Nn = 16384;   // rows of x
constexpr int Kk = 2048;    // rows of mu / std (output cols)
constexpr int Dd = 512;     // feature dim

__device__ __forceinline__ u16 f32_to_bf16_rne(float f) {
    unsigned int u = __float_as_uint(f);
    unsigned int lsb = (u >> 16) & 1u;
    u += 0x7FFFu + lsb;               // round to nearest even
    return (u16)(u >> 16);
}
__device__ __forceinline__ float bf16_to_f32(u16 h) {
    return __uint_as_float(((unsigned int)h) << 16);
}

// -------------------------- fp32 -> (hi, lo) bf16 ---------------------------
__global__ void split_bf16_kernel(const float* __restrict__ in,
                                  u16* __restrict__ hi, u16* __restrict__ lo,
                                  int n4) {
    int i = blockIdx.x * blockDim.x + threadIdx.x;
    if (i >= n4) return;
    float4 v = reinterpret_cast<const float4*>(in)[i];
    ushort4 h, l;
    h.x = f32_to_bf16_rne(v.x); l.x = f32_to_bf16_rne(v.x - bf16_to_f32(h.x));
    h.y = f32_to_bf16_rne(v.y); l.y = f32_to_bf16_rne(v.y - bf16_to_f32(h.y));
    h.z = f32_to_bf16_rne(v.z); l.z = f32_to_bf16_rne(v.z - bf16_to_f32(h.z));
    h.w = f32_to_bf16_rne(v.w); l.w = f32_to_bf16_rne(v.w - bf16_to_f32(h.w));
    reinterpret_cast<ushort4*>(hi)[i] = h;
    reinterpret_cast<ushort4*>(lo)[i] = l;
}

// ------------------------------ ||x_n||^2 -----------------------------------
__global__ void xsq_kernel(const float* __restrict__ x, float* __restrict__ xsq) {
    int row  = blockIdx.x * 8 + (threadIdx.x >> 5);   // one wave32 per row
    int lane = threadIdx.x & 31;
    const float* p = x + (size_t)row * Dd;
    float s = 0.f;
    for (int j = lane; j < Dd; j += 32) { float v = p[j]; s = fmaf(v, v, s); }
    #pragma unroll
    for (int off = 16; off; off >>= 1) s += __shfl_xor(s, off, 32);
    if (lane == 0) xsq[row] = s;
}

// ---------------------- per-k stats: inv_denom, beta ------------------------
__global__ void kstats_kernel(const float* __restrict__ mu,
                              const float* __restrict__ sd,
                              float* __restrict__ invd, float* __restrict__ beta) {
    int row  = blockIdx.x * 8 + (threadIdx.x >> 5);   // one wave32 per k
    int lane = threadIdx.x & 31;
    const float* pm = mu + (size_t)row * Dd;
    const float* ps = sd + (size_t)row * Dd;
    float s1 = 0.f, s2 = 0.f, m2 = 0.f;
    for (int j = lane; j < Dd; j += 32) {
        float sv = ps[j], mv = pm[j];
        s1 += sv;
        s2 = fmaf(sv, sv, s2);
        m2 = fmaf(mv, mv, m2);
    }
    #pragma unroll
    for (int off = 16; off; off >>= 1) {
        s1 += __shfl_xor(s1, off, 32);
        s2 += __shfl_xor(s2, off, 32);
        m2 += __shfl_xor(m2, off, 32);
    }
    if (lane == 0) {
        const float HALF_LOG_2PI = 0.91893853320467274178f;
        float id = 1.0f / (2.0f * s2 + 1e-8f);
        invd[row] = id;
        beta[row] = -m2 * id - __logf(s1) - HALF_LOG_2PI;
    }
}

// --------------------------- WMMA fragment load -----------------------------
// 16-bit A-matrix layout (ISA 7.12.2): lane L holds row M=L&15; for lanes 0-15
// VGPR0..3 = K 0..7, VGPR4..7 = K 16..23; lanes 16-31 get K+8. So per lane:
// two b128 chunks at element offsets (kOff) and (kOff+16), kOff = (L>>4)*8.
// B (K x 16) for row-major mu uses the identical pattern with M -> column.
__device__ __forceinline__ v16bf load_frag(const u16* __restrict__ p) {
    v8bf c0 = *reinterpret_cast<const v8bf*>(p);
    v8bf c1 = *reinterpret_cast<const v8bf*>(p + 16);
    return __builtin_shufflevector(c0, c1, 0, 1, 2, 3, 4, 5, 6, 7,
                                           8, 9, 10, 11, 12, 13, 14, 15);
}

__device__ __forceinline__ v8f wmma_bf16(v16bf a, v16bf b, v8f c) {
    return __builtin_amdgcn_wmma_f32_16x16x32_bf16(
        /*neg_a=*/false, a, /*neg_b=*/false, b,
        /*c_mod=*/(short)0, c, /*reuse_a=*/false, /*reuse_b=*/false);
}

// --------------------------- fused GEMM + epilogue --------------------------
// Block = 256 threads = 8 waves arranged 4(M) x 2(N); wave tile = 32x64 out.
// Block tile = 128(M) x 128(K-cols). Grid = (Kk/128, Nn/128).
__global__ __launch_bounds__(256)
void gmm_logpdf_kernel(const u16* __restrict__ xh, const u16* __restrict__ xl,
                       const u16* __restrict__ mh, const u16* __restrict__ ml,
                       const float* __restrict__ xsq,
                       const float* __restrict__ invd,
                       const float* __restrict__ beta,
                       float* __restrict__ out) {
    const int lane = threadIdx.x & 31;
    const int wid  = threadIdx.x >> 5;
    const int mW   = wid & 3;            // 0..3  (M direction within block)
    const int nW   = wid >> 2;           // 0..1  (N direction within block)
    const int row0 = blockIdx.y * 128 + mW * 32;   // x-row base of wave tile
    const int col0 = blockIdx.x * 128 + nW * 64;   // mu-row (out-col) base
    const int lr   = lane & 15;
    const int kOff = (lane >> 4) << 3;   // 0 or 8 (+16 for second chunk)

    v8f acc[2][4] = {};                  // [mt][nt] fp32 accumulators (32x64)

    const int ra0 = (row0 + lr)      * Dd;   // A rows, m-tile 0
    const int ra1 = (row0 + 16 + lr) * Dd;   // A rows, m-tile 1
    int rb[4];
    #pragma unroll
    for (int nt = 0; nt < 4; ++nt) rb[nt] = (col0 + nt * 16 + lr) * Dd;

    for (int kb = 0; kb < Dd; kb += 32) {
        const int o = kb + kOff;
        v16bf ah0 = load_frag(xh + ra0 + o);
        v16bf al0 = load_frag(xl + ra0 + o);
        v16bf ah1 = load_frag(xh + ra1 + o);
        v16bf al1 = load_frag(xl + ra1 + o);

        v16bf bh[4], bl[4];
        #pragma unroll
        for (int nt = 0; nt < 4; ++nt) {
            bh[nt] = load_frag(mh + rb[nt] + o);
            bl[nt] = load_frag(ml + rb[nt] + o);
        }

        // 3-product split accumulation: hi*hi + hi*lo + lo*hi
        #pragma unroll
        for (int nt = 0; nt < 4; ++nt) {
            acc[0][nt] = wmma_bf16(ah0, bh[nt], acc[0][nt]);
            acc[0][nt] = wmma_bf16(ah0, bl[nt], acc[0][nt]);
            acc[0][nt] = wmma_bf16(al0, bh[nt], acc[0][nt]);

            acc[1][nt] = wmma_bf16(ah1, bh[nt], acc[1][nt]);
            acc[1][nt] = wmma_bf16(ah1, bl[nt], acc[1][nt]);
            acc[1][nt] = wmma_bf16(al1, bh[nt], acc[1][nt]);
        }
    }

    // Epilogue: C/D layout — lane L holds column N = L&15; VGPR r holds row
    // M = r (lanes 0-15) or r+8 (lanes 16-31).
    #pragma unroll
    for (int nt = 0; nt < 4; ++nt) {
        const int kcol = col0 + nt * 16 + lr;
        const float id = invd[kcol];
        const float bt = beta[kcol];
        #pragma unroll
        for (int mt = 0; mt < 2; ++mt) {
            const int rbase = row0 + mt * 16 + ((lane >> 4) << 3);
            v8f c = acc[mt][nt];
            #pragma unroll
            for (int r = 0; r < 8; ++r) {
                const int row = rbase + r;
                float v = fmaf(2.0f * c[r] - xsq[row], id, bt);
                out[(size_t)row * Kk + kcol] = v;
            }
        }
    }
}

// ---------------------------------------------------------------------------
extern "C" void kernel_launch(void* const* d_in, const int* in_sizes, int n_in,
                              void* d_out, int out_size, void* d_ws, size_t ws_size,
                              hipStream_t stream) {
    (void)in_sizes; (void)n_in; (void)out_size; (void)ws_size;
    const float* x  = (const float*)d_in[0];   // [Nn, Dd]
    const float* mu = (const float*)d_in[1];   // [Kk, Dd]
    const float* sd = (const float*)d_in[2];   // [Kk, Dd]
    float* out = (float*)d_out;                // [Nn, Kk]

    char* ws = (char*)d_ws;
    u16*   xh   = (u16*)(ws);                                          // 16 MB
    u16*   xl   = (u16*)(ws + (size_t)Nn * Dd * 2);                    // 16 MB
    u16*   mh   = (u16*)(ws + (size_t)Nn * Dd * 4);                    //  2 MB
    u16*   ml   = (u16*)(ws + (size_t)Nn * Dd * 4 + (size_t)Kk * Dd * 2);
    float* xsq  = (float*)(ws + (size_t)Nn * Dd * 4 + (size_t)Kk * Dd * 4);
    float* invd = xsq + Nn;
    float* beta = invd + Kk;

    // 1) split fp32 -> bf16 hi/lo (x and mu)
    split_bf16_kernel<<<(Nn * Dd / 4 + 255) / 256, 256, 0, stream>>>(x, xh, xl, Nn * Dd / 4);
    split_bf16_kernel<<<(Kk * Dd / 4 + 255) / 256, 256, 0, stream>>>(mu, mh, ml, Kk * Dd / 4);

    // 2) per-row scalars
    xsq_kernel<<<Nn / 8, 256, 0, stream>>>(x, xsq);
    kstats_kernel<<<Kk / 8, 256, 0, stream>>>(mu, sd, invd, beta);

    // 3) fused split-bf16 WMMA GEMM + log-pdf epilogue
    dim3 grid(Kk / 128, Nn / 128);
    gmm_logpdf_kernel<<<grid, 256, 0, stream>>>(xh, xl, mh, ml, xsq, invd, beta, out);
}
